// customModel_85572928406083
// MI455X (gfx1250) — matrix operand order repaired and checked by reading, not compiled, not run on hardware
//
#include <hip/hip_runtime.h>
#include <hip/hip_bf16.h>

typedef __bf16 bf16_t;
typedef __attribute__((ext_vector_type(16))) __bf16 v16bf;
typedef __attribute__((ext_vector_type(8)))  __bf16 v8bf;
typedef __attribute__((ext_vector_type(8)))  float  v8f;

__device__ __forceinline__ float sigmoidf_(float x) {
    return 1.0f / (1.0f + __expf(-x));
}

// ---------------------------------------------------------------------------
// fc0: h0[n][j] = sigmoid(sum_i x[n][i]*W0[i][j] + b0[j]);  D_IN=3, D0=64
// K=3 is too small for WMMA; pure VALU, writes bf16.
// ---------------------------------------------------------------------------
__global__ void fc0_kernel(const float* __restrict__ x, const float* __restrict__ W0,
                           const float* __restrict__ b0, bf16_t* __restrict__ h0, int N) {
    int t = blockIdx.x * blockDim.x + threadIdx.x;
    if (t >= N * 64) return;
    int n = t >> 6, j = t & 63;
    float acc = b0[j];
    acc += x[n * 3 + 0] * W0[0 * 64 + j];
    acc += x[n * 3 + 1] * W0[1 * 64 + j];
    acc += x[n * 3 + 2] * W0[2 * 64 + j];
    h0[t] = (bf16_t)sigmoidf_(acc);
}

// ---------------------------------------------------------------------------
// Transpose f32 weight [din,dout] -> bf16 [dout,din] so the WMMA B-fragment
// (one column of B per lane, K contiguous) is a single contiguous 32B load.
// ---------------------------------------------------------------------------
__global__ void wtrans_kernel(const float* __restrict__ W, bf16_t* __restrict__ Wt,
                              int din, int dout) {
    int t = blockIdx.x * blockDim.x + threadIdx.x;
    if (t >= din * dout) return;
    int i = t / dout, j = t - i * dout;
    Wt[j * din + i] = (bf16_t)W[t];
}

// ---------------------------------------------------------------------------
// Gather + mean over K=16 neighbors, vectorized 8 features per thread:
//   agg[n][8g..8g+7] = (1/16) * sum_k h[idx[n][k]][8g..8g+7]
// Each neighbor row is fetched as b128 per lane (16 lanes of the same node
// cover the full 256B row for d=128); indices come in as four int4 loads.
// All h tensors (6.4-12.8MB) are L2-resident on the 192MB L2.
// ---------------------------------------------------------------------------
__global__ void aggregate_kernel(const bf16_t* __restrict__ h, const int* __restrict__ idx,
                                 bf16_t* __restrict__ agg, int N, int d) {
    int g8 = d >> 3;                       // feature groups of 8 per node
    int t = blockIdx.x * blockDim.x + threadIdx.x;
    if (t >= N * g8) return;
    int n = t / g8, g = t - n * g8;

    const int4* r4 = (const int4*)(idx + n * 16);
    int4 i0 = r4[0], i1 = r4[1], i2 = r4[2], i3 = r4[3];
    int nb[16] = { i0.x, i0.y, i0.z, i0.w,  i1.x, i1.y, i1.z, i1.w,
                   i2.x, i2.y, i2.z, i2.w,  i3.x, i3.y, i3.z, i3.w };

    const bf16_t* hb = h + (long)(g << 3);
    float acc[8] = {};
    #pragma unroll
    for (int k = 0; k < 16; ++k) {
        v8bf v = *(const v8bf*)(hb + (long)nb[k] * d);
        #pragma unroll
        for (int i = 0; i < 8; ++i) acc[i] += (float)v[i];
    }
    v8bf o;
    #pragma unroll
    for (int i = 0; i < 8; ++i) o[i] = (bf16_t)(acc[i] * 0.0625f);
    *(v8bf*)(agg + (long)t * 8) = o;
}

// ---------------------------------------------------------------------------
// Register-blocked WMMA GEMM + bias + sigmoid, bf16 in/out, fp32 accumulate.
// One wave32 owns a 16-row strip and ALL DOUT/16 column tiles:
//   - A fragments (DIN/32 of them, <=32 VGPRs) loaded ONCE per strip
//   - loop over column tiles: load B fragments (weights, WGP$-resident),
//     DIN/32 chained v_wmma_f32_16x16x32_bf16, then bias+sigmoid+store.
// Fragment layouts used here (wave32; contiguous K, same order in A and B):
//   A  lane l: row (l&15); a[0..15]=K=(l>>4)*16+{0..15}
//   B  lane l: col (l&15); b[0..15]=K=(l>>4)*16+{0..15}   (Bt is [DOUT,DIN])
//   D  vgpr v, lane l: D[v + 8*(l>>4)][l&15]
// ---------------------------------------------------------------------------
template<int DIN, int DOUT>
__global__ void gemm_wmma_rb_kernel(const bf16_t* __restrict__ A, const bf16_t* __restrict__ Bt,
                                    const float* __restrict__ bias, bf16_t* __restrict__ C,
                                    int M) {
    constexpr int KF = DIN / 32;           // k fragments (2 or 4)
    constexpr int TN = DOUT / 16;          // column tiles (8 or 16)
    int wave = (int)((blockIdx.x * blockDim.x + threadIdx.x) >> 5);
    int lane = threadIdx.x & 31;
    if (wave * 16 >= M) return;            // wave-uniform: EXEC all-1s at WMMA

    int half = lane & 15;
    int grp  = lane >> 4;

    // Load the strip's A fragments once.
    const bf16_t* arow = A + (long)(wave * 16 + half) * DIN;
    v16bf afrag[KF];
    #pragma unroll
    for (int kf = 0; kf < KF; ++kf) {
        v8bf lo = *(const v8bf*)(arow + kf * 32 + grp * 16);
        v8bf hi = *(const v8bf*)(arow + kf * 32 + grp * 16 + 8);
        afrag[kf] = __builtin_shufflevector(lo, hi, 0, 1, 2, 3, 4, 5, 6, 7,
                                                    8, 9, 10, 11, 12, 13, 14, 15);
    }

    const bf16_t* bbase = Bt + (long)half * DIN + grp * 16;
    #pragma unroll
    for (int nt = 0; nt < TN; ++nt) {
        v8f acc = {};
        #pragma unroll
        for (int kf = 0; kf < KF; ++kf) {
            v16bf b = *(const v16bf*)(bbase + (long)nt * 16 * DIN + kf * 32);
            acc = __builtin_amdgcn_wmma_f32_16x16x32_bf16(
                      /*neg_a=*/false, afrag[kf], /*neg_b=*/false, b,
                      /*c_mod=*/(short)0, acc, /*reuse_a=*/false, /*reuse_b=*/false);
        }
        float bs = bias[nt * 16 + half];
        bf16_t* cptr = C + (long)(wave * 16 + grp * 8) * DOUT + (nt * 16 + half);
        #pragma unroll
        for (int v = 0; v < 8; ++v) {
            cptr[(long)v * DOUT] = (bf16_t)sigmoidf_(acc[v] + bs);
        }
    }
}

// ---------------------------------------------------------------------------
// fc2: out[n] = dot(h[n, 0:256], Wf2) + bf2.  One wave32 per node; each lane
// loads 8 contiguous bf16 (one b128) + 8 f32 weights, then wave32 reduction.
// ---------------------------------------------------------------------------
__global__ void fc2_kernel(const bf16_t* __restrict__ h, const float* __restrict__ Wf2,
                           const float* __restrict__ bf2, float* __restrict__ out, int N) {
    int wave = (int)((blockIdx.x * blockDim.x + threadIdx.x) >> 5);
    int lane = threadIdx.x & 31;
    if (wave >= N) return;
    v8bf v = *(const v8bf*)(h + (long)wave * 256 + lane * 8);
    const float* w = Wf2 + lane * 8;
    float acc = 0.0f;
    #pragma unroll
    for (int i = 0; i < 8; ++i) acc += (float)v[i] * w[i];
    #pragma unroll
    for (int off = 16; off > 0; off >>= 1)
        acc += __shfl_down(acc, off, 32);
    if (lane == 0) out[wave] = acc + bf2[0];
}

// ---------------------------------------------------------------------------
extern "C" void kernel_launch(void* const* d_in, const int* in_sizes, int n_in,
                              void* d_out, int out_size, void* d_ws, size_t ws_size,
                              hipStream_t stream) {
    const float* x   = (const float*)d_in[0];
    const int*   idx = (const int*)d_in[1];
    const float* W0  = (const float*)d_in[2];
    const float* b0  = (const float*)d_in[3];
    const float* W1  = (const float*)d_in[4];
    const float* b1  = (const float*)d_in[5];
    const float* W2  = (const float*)d_in[6];
    const float* b2  = (const float*)d_in[7];
    const float* W3  = (const float*)d_in[8];
    const float* b3  = (const float*)d_in[9];
    const float* Wf1 = (const float*)d_in[10];
    const float* bf1 = (const float*)d_in[11];
    const float* Wf2 = (const float*)d_in[12];
    const float* bf2 = (const float*)d_in[13];
    float* out = (float*)d_out;

    const int N = in_sizes[0] / 3;      // x is [N,3]

    // workspace carve-out (256B aligned slabs)
    char* ws = (char*)d_ws;
    auto carve = [&](size_t bytes) -> char* {
        char* p = ws;
        ws += (bytes + 255) & ~(size_t)255;
        return p;
    };
    bf16_t* hA   = (bf16_t*)carve((size_t)N * 256 * sizeof(bf16_t));
    bf16_t* hB   = (bf16_t*)carve((size_t)N * 256 * sizeof(bf16_t));
    bf16_t* agg  = (bf16_t*)carve((size_t)N * 128 * sizeof(bf16_t));
    bf16_t* w1t  = (bf16_t*)carve(64  * 128 * sizeof(bf16_t));
    bf16_t* w2t  = (bf16_t*)carve(128 * 128 * sizeof(bf16_t));
    bf16_t* w3t  = (bf16_t*)carve(128 * 128 * sizeof(bf16_t));
    bf16_t* wf1t = (bf16_t*)carve(128 * 256 * sizeof(bf16_t));

    // transpose + bf16-convert weights (tiny)
    wtrans_kernel<<<(64 * 128  + 255) / 256, 256, 0, stream>>>(W1,  w1t,  64, 128);
    wtrans_kernel<<<(128 * 128 + 255) / 256, 256, 0, stream>>>(W2,  w2t,  128, 128);
    wtrans_kernel<<<(128 * 128 + 255) / 256, 256, 0, stream>>>(W3,  w3t,  128, 128);
    wtrans_kernel<<<(128 * 256 + 255) / 256, 256, 0, stream>>>(Wf1, wf1t, 128, 256);

    // fc0
    fc0_kernel<<<(N * 64 + 255) / 256, 256, 0, stream>>>(x, W0, b0, hA, N);

    const int mtiles = (N + 15) / 16;          // one wave per 16-row strip
    const int gblocks = (mtiles + 7) / 8;      // 8 waves / 256-thread block

    // conv1: gather-mean(hA, d=64) -> agg; agg @ W1 -> hB [N,128]
    aggregate_kernel<<<(N * 8  + 255) / 256, 256, 0, stream>>>(hA, idx, agg, N, 64);
    gemm_wmma_rb_kernel<64, 128><<<gblocks, 256, 0, stream>>>(agg, w1t, b1, hB, N);

    // conv2: hB -> agg; agg @ W2 -> hA [N,128]
    aggregate_kernel<<<(N * 16 + 255) / 256, 256, 0, stream>>>(hB, idx, agg, N, 128);
    gemm_wmma_rb_kernel<128, 128><<<gblocks, 256, 0, stream>>>(agg, w2t, b2, hA, N);

    // conv3: hA -> agg; agg @ W3 -> hB [N,128]
    aggregate_kernel<<<(N * 16 + 255) / 256, 256, 0, stream>>>(hA, idx, agg, N, 128);
    gemm_wmma_rb_kernel<128, 128><<<gblocks, 256, 0, stream>>>(agg, w3t, b3, hB, N);

    // fc1: hB [N,128] @ Wf1 -> hA [N,256]
    gemm_wmma_rb_kernel<128, 256><<<gblocks, 256, 0, stream>>>(hB, wf1t, bf1, hA, N);

    // fc2: hA [N,256] . Wf2 + bf2 -> out [N]
    fc2_kernel<<<(N + 7) / 8, 256, 0, stream>>>(hA, Wf2, bf2, out, N);
}
